// GatedCrossAttention_18305150615720
// MI455X (gfx1250) — compile-verified, matrix operand order checked by reading, not run on hardware
//
#include <hip/hip_runtime.h>
#include <hip/hip_bf16.h>

// ---------------------------------------------------------------------------
// GatedCrossAttention for MI455X (gfx1250): wave32, WMMA bf16, async LDS.
// Dead code in the reference (Wv/bv, Wg/bg, gate) is skipped: v = g = k, so
// the gated value reduces to kproj^2.
// ---------------------------------------------------------------------------

typedef __attribute__((ext_vector_type(16))) __bf16 v16bf;
typedef __attribute__((ext_vector_type(8)))  __bf16 v8bf;
typedef __attribute__((ext_vector_type(8)))  float  v8f;
typedef __attribute__((ext_vector_type(4)))  float  f32x4;

#define CDIM 1024            // embed dim = K = N of all GEMMs

// LDS geometry for the fused kernel (all byte offsets; dynamic LDS base == 0)
#define FSTRIDE   516        // fp32 slab row stride in floats (2064 B: 16B mod 256B -> conflict-free)
#define KSTRIDE   1032       // bf16 kproj row stride in elems (2064 B)
#define FSLAB_OFF 0u
#define KBUF_OFF  132096u    // 64 * 516 * 4
#define LOGIT_OFF 264192u    // KBUF_OFF + 64 * 1032 * 2
#define PROB_OFF  268288u
#define SMEM_SZ   272384u

__device__ __forceinline__ v8f zero8() {
  v8f z = {0.f, 0.f, 0.f, 0.f, 0.f, 0.f, 0.f, 0.f};
  return z;
}

__device__ __forceinline__ v8f wmma_bf16(v16bf a, v16bf b, v8f c) {
  return __builtin_amdgcn_wmma_f32_16x16x32_bf16(
      false, a, false, b, (short)0, c, false, false);
}

// Async copy global -> LDS (gfx1250 GLOBAL_LOAD_ASYNC_TO_LDS_B128, ASYNCcnt).
__device__ __forceinline__ void async_g2l_b128(unsigned lds_byte_off,
                                               const void* gaddr) {
  asm volatile("global_load_async_to_lds_b128 %0, %1, off"
               :: "v"(lds_byte_off), "v"(gaddr) : "memory");
}
__device__ __forceinline__ void wait_async0() {
  asm volatile("s_wait_asynccnt 0x0" ::: "memory");
}

// A/B fragment for v_wmma_f32_16x16x32_bf16 from row-major [rows x ld] bf16,
// contracting over the last dim (ISA 7.12.2, 16-bit A 16x32 layout):
//   lanes 0-15 : row = base+lane,    K in {k0..k0+7,  k0+16..k0+23}
//   lanes 16-31: row = base+lane-16, K in {k0+8..k0+15, k0+24..k0+31}
__device__ __forceinline__ v16bf frag_bf16(const __bf16* base, int ld,
                                           int row0, int k0, int lane) {
  const int r    = row0 + (lane & 15);
  const int koff = (lane >> 4) * 8;
  const v8bf lo = *(const v8bf*)(base + (size_t)r * ld + k0 + koff);
  const v8bf hi = *(const v8bf*)(base + (size_t)r * ld + k0 + 16 + koff);
  v16bf a;
#pragma unroll
  for (int i = 0; i < 8; ++i) { a[i] = lo[i]; a[i + 8] = hi[i]; }
  return a;
}

// Same fragment sourced from fp32 (global or LDS), converted bf16 in regs.
__device__ __forceinline__ v16bf frag_f32(const float* base, int ld,
                                          int row0, int k0, int lane) {
  const int r    = row0 + (lane & 15);
  const int koff = (lane >> 4) * 8;
  const f32x4* p = (const f32x4*)(base + (size_t)r * ld + k0 + koff);
  const f32x4* q = (const f32x4*)(base + (size_t)r * ld + k0 + 16 + koff);
  f32x4 x0 = p[0], x1 = p[1], x2 = q[0], x3 = q[1];
  v16bf a;
#pragma unroll
  for (int i = 0; i < 4; ++i) {
    a[i]      = (__bf16)x0[i];
    a[i + 4]  = (__bf16)x1[i];
    a[i + 8]  = (__bf16)x2[i];
    a[i + 12] = (__bf16)x3[i];
  }
  return a;
}

// ---------------------------------------------------------------------------
// fp32 -> bf16 weight conversion
// ---------------------------------------------------------------------------
__global__ void cvt_w_k(const float* __restrict__ src, __bf16* __restrict__ dst,
                        int n) {
  int i = blockIdx.x * 256 + threadIdx.x;
  if (i < n) dst[i] = (__bf16)src[i];
}

// ---------------------------------------------------------------------------
// GEMM: Out[rows x 1024] = A[rows x 1024] @ W^T + bias.
// W row-major [1024 out x 1024 in] bf16, so B fragments load like A fragments.
// 4 waves / block; each wave owns a 64x64 tile (4Mx4N, 16 WMMAs per K-step);
// the 4 waves cover adjacent column strips so they share A rows through L1.
// grid.x = strip group (4), grid.y = row tile (rows/64).
// ---------------------------------------------------------------------------
template <bool ABF16, bool OBF16>
__global__ __launch_bounds__(128)
void gemm_bias_k(const void* __restrict__ Av, const __bf16* __restrict__ W,
                 const float* __restrict__ bias, void* __restrict__ Outv) {
  const int lane = threadIdx.x & 31;
  const int wave = threadIdx.x >> 5;
  const int col0 = (blockIdx.x * 4 + wave) * 64;
  const int row0 = blockIdx.y * 64;

  v8f acc[4][4];                       // [mt][nt]
#pragma unroll
  for (int i = 0; i < 4; ++i)
#pragma unroll
    for (int j = 0; j < 4; ++j) acc[i][j] = zero8();

  for (int k0 = 0; k0 < CDIM; k0 += 32) {
    __builtin_prefetch(W + (size_t)(col0 + (lane & 15)) * CDIM + k0 + 128, 0, 0);
    v16bf a[4];
#pragma unroll
    for (int mt = 0; mt < 4; ++mt) {
      if constexpr (ABF16)
        a[mt] = frag_bf16((const __bf16*)Av, CDIM, row0 + mt * 16, k0, lane);
      else
        a[mt] = frag_f32((const float*)Av, CDIM, row0 + mt * 16, k0, lane);
    }
#pragma unroll
    for (int nt = 0; nt < 4; ++nt) {
      v16bf b = frag_bf16(W, CDIM, col0 + nt * 16, k0, lane);
#pragma unroll
      for (int mt = 0; mt < 4; ++mt)
        acc[mt][nt] = wmma_bf16(a[mt], b, acc[mt][nt]);
    }
  }

  // D layout: lanes 0-15 vgpr r -> M=r; lanes 16-31 -> M=r+8; N = lane&15.
  const int c_lo = lane & 15;
  const int radd = (lane >> 4) * 8;
#pragma unroll
  for (int nt = 0; nt < 4; ++nt) {
    const int col = col0 + nt * 16 + c_lo;
    const float bv = bias[col];
#pragma unroll
    for (int mt = 0; mt < 4; ++mt)
#pragma unroll
      for (int r = 0; r < 8; ++r) {
        const size_t idx = (size_t)(row0 + mt * 16 + radd + r) * CDIM + col;
        const float v = acc[mt][nt][r] + bv;
        if constexpr (OBF16) ((__bf16*)Outv)[idx] = (__bf16)v;
        else                 ((float*)Outv)[idx]  = v;
      }
  }
}

// ---------------------------------------------------------------------------
// Fused K-projection + attention. 512 threads (16 waves) per (b, 16-row tile):
//   for khalf in {0,1}:
//     async-stage key slab [64 x 512] fp32 -> LDS  (global_load_async_to_lds)
//     WMMA kp += slab @ Wk^T  (wave owns 4 col tiles x 4 M-tiles, accums live)
//   kp + bk -> bf16 LDS; logits -> softmax(M=4) -> pre = sum_m p * kp^2 -> ws
// The 256 MB kproj tensor never touches HBM.
// ---------------------------------------------------------------------------
__global__ __launch_bounds__(512)
void kproj_attn_k(const float* __restrict__ key, const __bf16* __restrict__ Q,
                  const __bf16* __restrict__ Wk, const float* __restrict__ bk,
                  __bf16* __restrict__ att) {
  extern __shared__ char smem[];
  float*  fslab  = (float*)(smem + FSLAB_OFF);   // [64][FSTRIDE] fp32 key half-slab
  __bf16* kbuf   = (__bf16*)(smem + KBUF_OFF);   // [64][KSTRIDE] bf16 kproj
  float*  logits = (float*)(smem + LOGIT_OFF);   // [16 n][4 m][16 h]
  float*  probs  = (float*)(smem + PROB_OFF);

  const int tid  = threadIdx.x;
  const int wave = tid >> 5;
  const int lane = tid & 31;
  const int b    = blockIdx.y;
  const int nt16 = blockIdx.x;
  const size_t keyRow0 = ((size_t)b * 4096 + (size_t)nt16 * 16) * 4;

  v8f acc[4][4];                       // [nt0][mt]
#pragma unroll
  for (int i = 0; i < 4; ++i)
#pragma unroll
    for (int j = 0; j < 4; ++j) acc[i][j] = zero8();

  for (int kh = 0; kh < 2; ++kh) {
    // ---- async stage: 64 rows x 512 floats = 8192 x b128 chunks ----
    for (int i = tid; i < 8192; i += 512) {
      const int r = i >> 7, c = i & 127;             // 128 chunks per row
      const void* g = key + (keyRow0 + r) * CDIM + kh * 512 + c * 4;
      async_g2l_b128(FSLAB_OFF + (unsigned)(r * (FSTRIDE * 4) + c * 16), g);
    }
    wait_async0();
    __syncthreads();

    // ---- GEMM over this K half ----
    for (int k0 = 0; k0 < 512; k0 += 32) {
      v16bf a[4];
#pragma unroll
      for (int mt = 0; mt < 4; ++mt)
        a[mt] = frag_f32(fslab, FSTRIDE, mt * 16, k0, lane);
#pragma unroll
      for (int nt0 = 0; nt0 < 4; ++nt0) {
        const int col0 = (wave * 4 + nt0) * 16;
        v16bf bf_ = frag_bf16(Wk, CDIM, col0, kh * 512 + k0, lane);
#pragma unroll
        for (int mt = 0; mt < 4; ++mt)
          acc[nt0][mt] = wmma_bf16(a[mt], bf_, acc[nt0][mt]);
      }
    }
    __syncthreads();   // all reads of fslab done before restaging
  }

  // ---- epilogue: +bias, store kproj tiles to LDS as bf16 ----
  {
    const int c_lo = lane & 15;
    const int radd = (lane >> 4) * 8;
#pragma unroll
    for (int nt0 = 0; nt0 < 4; ++nt0) {
      const int col = (wave * 4 + nt0) * 16 + c_lo;
      const float bv = bk[col];
#pragma unroll
      for (int mt = 0; mt < 4; ++mt)
#pragma unroll
        for (int r = 0; r < 8; ++r)
          kbuf[(mt * 16 + radd + r) * KSTRIDE + col] =
              (__bf16)(acc[nt0][mt][r] + bv);
    }
  }
  __syncthreads();

  // ---- logits[n,m,h] = SCALE * <q[n,h,:], kp[n,m,h,:]> over 64 channels ----
  const __bf16* qbase = Q + ((size_t)b * 4096 + (size_t)nt16 * 16) * CDIM;
  for (int t = tid; t < 1024; t += 512) {
    const int h = t & 15, m = (t >> 4) & 3, n = t >> 6;
    const __bf16* kr = kbuf + (n * 4 + m) * KSTRIDE + h * 64;
    const __bf16* qp = qbase + (size_t)n * CDIM + h * 64;
    float s = 0.f;
#pragma unroll 8
    for (int i = 0; i < 64; ++i) s += (float)qp[i] * (float)kr[i];
    logits[t] = s * 0.125f;                    // SCALE = 64^-0.5
  }
  __syncthreads();

  // ---- softmax over m (4) per (n,h): 256 tasks ----
  if (tid < 256) {
    const int h = tid & 15, n = tid >> 4;
    float l[4];
#pragma unroll
    for (int m = 0; m < 4; ++m) l[m] = logits[n * 64 + m * 16 + h];
    float mx = fmaxf(fmaxf(l[0], l[1]), fmaxf(l[2], l[3]));
    float e[4], s = 0.f;
#pragma unroll
    for (int m = 0; m < 4; ++m) { e[m] = __expf(l[m] - mx); s += e[m]; }
    const float inv = 1.f / s;
#pragma unroll
    for (int m = 0; m < 4; ++m) probs[n * 64 + m * 16 + h] = e[m] * inv;
  }
  __syncthreads();

  // ---- pre[n,c] = sum_m p * kp^2  (v*g == k*k); 512 threads, 32 ch each ----
  {
    const int n = tid >> 5, rest = tid & 31;
    const int h = rest >> 1, jb = (rest & 1) * 32;
    float p[4];
#pragma unroll
    for (int m = 0; m < 4; ++m) p[m] = probs[n * 64 + m * 16 + h];
    __bf16* orow =
        att + ((size_t)b * 4096 + (size_t)nt16 * 16 + n) * CDIM + h * 64 + jb;
    const __bf16* k0p = kbuf + (n * 4 + 0) * KSTRIDE + h * 64 + jb;
    const __bf16* k1p = kbuf + (n * 4 + 1) * KSTRIDE + h * 64 + jb;
    const __bf16* k2p = kbuf + (n * 4 + 2) * KSTRIDE + h * 64 + jb;
    const __bf16* k3p = kbuf + (n * 4 + 3) * KSTRIDE + h * 64 + jb;
#pragma unroll 8
    for (int j = 0; j < 32; ++j) {
      const float v0 = (float)k0p[j], v1 = (float)k1p[j];
      const float v2 = (float)k2p[j], v3 = (float)k3p[j];
      orow[j] = (__bf16)(p[0] * v0 * v0 + p[1] * v1 * v1 +
                         p[2] * v2 * v2 + p[3] * v3 * v3);
    }
  }
}

// ---------------------------------------------------------------------------
// Host side
// ---------------------------------------------------------------------------
extern "C" void kernel_launch(void* const* d_in, const int* in_sizes, int n_in,
                              void* d_out, int out_size, void* d_ws, size_t ws_size,
                              hipStream_t stream) {
  (void)in_sizes; (void)n_in; (void)out_size; (void)ws_size;
  const float* query = (const float*)d_in[0];   // [4,4096,1024]
  const float* key   = (const float*)d_in[1];   // [4,4096,4,1024]
  const float* Wq = (const float*)d_in[3];
  const float* bq = (const float*)d_in[4];
  const float* Wk = (const float*)d_in[5];
  const float* bk = (const float*)d_in[6];
  const float* Wo = (const float*)d_in[11];
  const float* bo = (const float*)d_in[12];
  float* out = (float*)d_out;                   // [4,4096,1024] fp32

  const size_t ROWS = (size_t)4 * 4096;         // 16384
  const size_t WSZ  = (size_t)CDIM * CDIM;

  // workspace (≈70 MB): Wq|Wk|Wo bf16, Q bf16, pre bf16
  char* ws = (char*)d_ws;
  __bf16* wqb  = (__bf16*)ws;
  __bf16* wkb  = wqb + WSZ;
  __bf16* wob  = wkb + WSZ;
  __bf16* Qbuf = wob + WSZ;                     // [16384 x 1024] bf16
  __bf16* preb = Qbuf + ROWS * CDIM;            // [16384 x 1024] bf16

  cvt_w_k<<<dim3((WSZ + 255) / 256), 256, 0, stream>>>(Wq, wqb, (int)WSZ);
  cvt_w_k<<<dim3((WSZ + 255) / 256), 256, 0, stream>>>(Wk, wkb, (int)WSZ);
  cvt_w_k<<<dim3((WSZ + 255) / 256), 256, 0, stream>>>(Wo, wob, (int)WSZ);

  // 1) Q projection (fp32 A converted on the fly, bf16 out)
  gemm_bias_k<false, true><<<dim3(4, ROWS / 64), 128, 0, stream>>>(query, wqb, bq, Qbuf);

  // 2) fused K projection + attention (async LDS staging, kp stays in LDS)
  kproj_attn_k<<<dim3(4096 / 16, 4), 512, SMEM_SZ, stream>>>(key, Qbuf, wkb, bk, preb);

  // 3) output projection (bf16 A, fp32 out)
  gemm_bias_k<true, false><<<dim3(4, ROWS / 64), 128, 0, stream>>>(preb, wob, bo, out);
}
